// PNA_1838246002977
// MI455X (gfx1250) — matrix operand order repaired and checked by reading, not compiled, not run on hardware
//
#include <hip/hip_runtime.h>
#include <hip/hip_bf16.h>
#include <math.h>

#define N_NODES 50000
#define N_EDGES 800000
#define FDIM 64
#define TOW 2
#define NLAYER 2
#define OUTD 2

typedef __attribute__((ext_vector_type(16))) __bf16 v16bf;
typedef __attribute__((ext_vector_type(8)))  __bf16 v8bf;
typedef __attribute__((ext_vector_type(8)))  float  v8f;
typedef __attribute__((ext_vector_type(4)))  unsigned int v4u;
typedef __attribute__((ext_vector_type(8)))  int v8i;
typedef __attribute__((ext_vector_type(4)))  int v4i;

__device__ __forceinline__ unsigned short f2bf(float f) {
    unsigned u = __float_as_uint(f);
    unsigned r = u + 0x7FFFu + ((u >> 16) & 1u);
    return (unsigned short)(r >> 16);
}

__device__ __forceinline__ v16bf cat16(v8bf lo, v8bf hi) {
    return __builtin_shufflevector(lo, hi, 0,1,2,3,4,5,6,7,8,9,10,11,12,13,14,15);
}

__device__ __forceinline__ v8f splat8(float x) {
    v8f v;
    #pragma unroll
    for (int i = 0; i < 8; i++) v[i] = x;
    return v;
}

// Native float min/max atomics (GLOBAL_ATOMIC_MIN/MAX_NUM_F32).
__device__ __forceinline__ void atomicMaxF(float* a, float v) {
    __hip_atomic_fetch_max(a, v, __ATOMIC_RELAXED, __HIP_MEMORY_SCOPE_AGENT);
}
__device__ __forceinline__ void atomicMinF(float* a, float v) {
    __hip_atomic_fetch_min(a, v, __ATOMIC_RELAXED, __HIP_MEMORY_SCOPE_AGENT);
}

// ---------------- small utility kernels ----------------

__global__ void fill_k(float* p, float v, int n) {
    int i = blockIdx.x * blockDim.x + threadIdx.x;
    if (i < n) p[i] = v;
}

__global__ void convert_bf16_k(unsigned short* dst, const float* src, int n) {
    int i = blockIdx.x * blockDim.x + threadIdx.x;
    if (i < n) dst[i] = f2bf(src[i]);
}

// dst[n*K + k] = bf16(src[k*Nn + n])   (transpose to [n][k] for WMMA B frags)
__global__ void transpose_bf16_k(unsigned short* dst, const float* src, int K, int Nn) {
    int i = blockIdx.x * blockDim.x + threadIdx.x;
    if (i >= K * Nn) return;
    int n = i / K, k = i % K;
    dst[n * K + k] = f2bf(src[k * Nn + n]);
}

__global__ void avg_log_k(const float* deg, float* scalars) {
    __shared__ float s1[64], s2[64];
    int i = threadIdx.x;
    s1[i] = logf((float)i + 1.f) * deg[i];
    s2[i] = deg[i];
    __syncthreads();
    if (i == 0) {
        float a = 0.f, b = 0.f;
        for (int k = 0; k < 64; k++) { a += s1[k]; b += s2[k]; }
        scalars[0] = a / b;
    }
}

__global__ void count_k(const int* tgt, float* cnt, int e) {
    int i = blockIdx.x * blockDim.x + threadIdx.x;
    if (i < e) atomicAdd(&cnt[tgt[i]], 1.f);
}

__global__ void init_acc_k(float* sum, float* sum2, float* mn, float* mx, int n) {
    int i = blockIdx.x * blockDim.x + threadIdx.x;
    if (i >= n) return;
    sum[i] = 0.f; sum2[i] = 0.f; mn[i] = 3.402823466e38f; mx[i] = -3.402823466e38f;
}

// ---------------- edge WMMA kernel ----------------
// One wave handles 16 edges. A = [16 edges x 128 feats] bf16 gathered as
// cat(h[tgt], h[src]); B = W_pre^T staged into LDS via the Tensor Data Mover
// (32KB, reused by all 4 waves). Four independent accumulator chains are
// interleaved (K-chunk outer, n-tile inner) to hide WMMA->WMMA RAW hazards.
__global__ void __launch_bounds__(128)
pna_edge_k(const unsigned short* __restrict__ hb,
           const int* __restrict__ src, const int* __restrict__ tgt,
           const unsigned short* __restrict__ wpreT,  // [2][64][128] bf16, 32KB
           const float* __restrict__ bpre,            // [2][64]
           float* __restrict__ sum, float* __restrict__ sum2,
           float* __restrict__ mn,  float* __restrict__ mx, int e) {
    __shared__ unsigned short sW[TOW * 64 * 128];  // sole LDS object -> offset 0

    // ---- stage W_pre^T into LDS ----
#if __has_builtin(__builtin_amdgcn_tensor_load_to_lds) && __has_builtin(__builtin_amdgcn_s_wait_tensorcnt)
    if ((threadIdx.x >> 5) == 0) {
        unsigned long long ga = (unsigned long long)wpreT;
        // D# group0: count=1 | lds_addr=0 | global_addr | type=2 ("image")
        v4u g0;
        g0[0] = 1u;
        g0[1] = 0u;  // LDS byte address of sW (only LDS allocation)
        g0[2] = (unsigned)(ga & 0xFFFFFFFFu);
        g0[3] = (unsigned)((ga >> 32) & 0x1FFFFFFu) | (2u << 30);
        // D# group1: data_size=2B; tensor 16384x1; tile 16384x1; stride 16384
        v8i g1;
        g1[0] = (int)(1u << 16);        // data_size = 1 (2 bytes)
        g1[1] = (int)(16384u << 16);    // tensor_dim0[15:0] in bits[63:48]
        g1[2] = (int)(1u << 16);        // tensor_dim1[15:0] in bits[95:80]
        g1[3] = (int)(16384u << 16);    // tile_dim0 in bits[127:112]
        g1[4] = 1;                      // tile_dim1 = 1
        g1[5] = 16384;                  // tensor_dim0_stride
        g1[6] = 0;
        g1[7] = 0;
        v4i z4 = {0, 0, 0, 0};
        v8i z8 = {0, 0, 0, 0, 0, 0, 0, 0};
        __builtin_amdgcn_tensor_load_to_lds(g0, g1, z4, z4, z8, 0);
        __builtin_amdgcn_s_wait_tensorcnt(0);
    }
#else
    for (int i = threadIdx.x; i < TOW * 64 * 128 / 8; i += 128)
        ((v8bf*)sW)[i] = ((const v8bf*)wpreT)[i];
#endif
    __syncthreads();

    int wave = threadIdx.x >> 5;
    int L    = threadIdx.x & 31;
    int tile = blockIdx.x * 4 + wave;
    int base = tile * 16;
    if (base >= e) return;
    int m = L & 15, half = L >> 4;

    int eg = base + m;
    int sn = src[eg], tn = tgt[eg];

    // A fragments: chunk c covers K = 32c..32c+31; c<2 -> h[tgt], c>=2 -> h[src]
    v16bf a[4];
    #pragma unroll
    for (int c = 0; c < 4; c++) {
        const unsigned short* row = ((c < 2) ? (hb + (size_t)tn * FDIM)
                                             : (hb + (size_t)sn * FDIM)) + (c & 1) * 32;
        v8bf lo = *reinterpret_cast<const v8bf*>(row + 8 * half);
        v8bf hi = *reinterpret_cast<const v8bf*>(row + 16 + 8 * half);
        a[c] = cat16(lo, hi);
    }

    int tg[8];
    #pragma unroll
    for (int r = 0; r < 8; r++) tg[r] = tgt[base + 8 * half + r];

    #pragma unroll
    for (int t = 0; t < TOW; t++) {
        v8f acc[4];
        #pragma unroll
        for (int j = 0; j < 4; j++) acc[j] = splat8(bpre[t * 64 + 16 * j + m]);

        // K-chunk outer, n-tile inner: consecutive WMMAs hit different accs.
        #pragma unroll
        for (int c = 0; c < 4; c++) {
            #pragma unroll
            for (int j = 0; j < 4; j++) {
                const unsigned short* bp = sW + (t * 64 + 16 * j + m) * 128 + 32 * c + 8 * half;
                v16bf b = cat16(*reinterpret_cast<const v8bf*>(bp),
                                *reinterpret_cast<const v8bf*>(bp + 16));
                acc[j] = __builtin_amdgcn_wmma_f32_16x16x32_bf16(false, a[c], false, b,
                                                                 (short)0, acc[j], false, false);
            }
        }

        #pragma unroll
        for (int j = 0; j < 4; j++) {
            unsigned ncol = (unsigned)(t * FDIM + 16 * j + m);
            #pragma unroll
            for (int r = 0; r < 8; r++) {
                float v = acc[j][r];
                unsigned idx = (unsigned)tg[r] * (TOW * FDIM) + ncol;
                atomicAdd(sum  + idx, v);
                atomicAdd(sum2 + idx, v * v);
                atomicMinF(mn + idx, v);
                atomicMaxF(mx + idx, v);
            }
        }
    }
}

// ---------------- finalize + post_nn WMMA kernel ----------------
// One wave (one block) per 16-node tile: build the [16 x 832] bf16 PNA
// feature tile in LDS, then WMMA against W_post^T -> pb [N,64] bf16.
__global__ void __launch_bounds__(32)
pna_finalize_k(const float* __restrict__ h, const float* __restrict__ cnt,
               const float* __restrict__ scalars,
               const float* __restrict__ sum, const float* __restrict__ sum2,
               const float* __restrict__ mn,  const float* __restrict__ mx,
               const unsigned short* __restrict__ wpostT, // [2][32][832]
               const float* __restrict__ bpost,           // [2][32]
               unsigned short* __restrict__ pb) {
    __shared__ unsigned short sA[16 * 832];
    int g0 = blockIdx.x * 16;
    int L = threadIdx.x, m = L & 15, half = L >> 4;
    float avg_log = scalars[0];

    for (int t = 0; t < TOW; t++) {
        for (int nd = 0; nd < 16; nd++) {
            int g = g0 + nd;
            float c  = cnt[g];
            float cs = fmaxf(c, 1.f);
            float logd = logf(cs + 1.f);
            float amp = logd / avg_log, att = avg_log / logd;
            bool has = c > 0.f;
            unsigned gb = (unsigned)(g * TOW + t) * FDIM;
            const float* su  = sum  + gb;
            const float* su2 = sum2 + gb;
            const float* mnp = mn   + gb;
            const float* mxp = mx   + gb;
            for (int f = L; f < 832; f += 32) {
                float v;
                if (f < 64) {
                    v = h[(unsigned)g * FDIM + f];
                } else {
                    int idx = f - 64;
                    int grp = idx >> 8;          // 0: raw, 1: *amp, 2: *att
                    int w = idx & 255;
                    int st = w >> 6, col = w & 63; // 0 mean, 1 min, 2 max, 3 std
                    float mean = su[col] / cs;
                    float val;
                    if (st == 0)      val = mean;
                    else if (st == 1) val = has ? mnp[col] : 0.f;
                    else if (st == 2) val = has ? mxp[col] : 0.f;
                    else {
                        float d = su2[col] / cs - mean * mean;
                        d = d > 0.f ? d : 0.f;
                        val = sqrtf(d + 1e-5f);
                    }
                    v = (grp == 0) ? val : ((grp == 1) ? val * amp : val * att);
                }
                sA[nd * 832 + f] = f2bf(v);
            }
        }
        __syncthreads();

        // Two interleaved accumulator chains (n-tiles j=0,1) hide WMMA hazards.
        v8f acc0 = splat8(bpost[t * 32 + m]);
        v8f acc1 = splat8(bpost[t * 32 + 16 + m]);
        for (int c = 0; c < 26; c++) {
            const unsigned short* row = sA + m * 832 + 32 * c;
            v16bf a = cat16(*reinterpret_cast<const v8bf*>(row + 8 * half),
                            *reinterpret_cast<const v8bf*>(row + 16 + 8 * half));
            const unsigned short* bp0 = wpostT + (unsigned)(t * 32 + m) * 832
                                      + 32 * c + 8 * half;
            const unsigned short* bp1 = wpostT + (unsigned)(t * 32 + 16 + m) * 832
                                      + 32 * c + 8 * half;
            v16bf b0 = cat16(*reinterpret_cast<const v8bf*>(bp0),
                             *reinterpret_cast<const v8bf*>(bp0 + 16));
            v16bf b1 = cat16(*reinterpret_cast<const v8bf*>(bp1),
                             *reinterpret_cast<const v8bf*>(bp1 + 16));
            acc0 = __builtin_amdgcn_wmma_f32_16x16x32_bf16(false, a, false, b0,
                                                           (short)0, acc0, false, false);
            acc1 = __builtin_amdgcn_wmma_f32_16x16x32_bf16(false, a, false, b1,
                                                           (short)0, acc1, false, false);
        }
        #pragma unroll
        for (int r = 0; r < 8; r++) {
            unsigned g = (unsigned)(g0 + r + 8 * half);
            pb[g * FDIM + t * 32 + m]      = f2bf(acc0[r]);
            pb[g * FDIM + t * 32 + 16 + m] = f2bf(acc1[r]);
        }
        __syncthreads();
    }
}

// ---------------- generic [M,64] x [64,64] WMMA GEMM ----------------
__global__ void __launch_bounds__(128)
gemm64_wmma_k(const unsigned short* __restrict__ A,   // [M,64] bf16
              const unsigned short* __restrict__ Bt,  // [64,64] bf16, [n][k]
              const float* __restrict__ bias,         // [64] or null
              float* __restrict__ Cf,                  // [M,64] or null
              unsigned short* __restrict__ Cb,         // [M,64] bf16 or null
              int M, int relu) {
    int wave = threadIdx.x >> 5;
    int L    = threadIdx.x & 31;
    int tile = blockIdx.x * 4 + wave;
    int r0 = tile * 16;
    if (r0 >= M) return;
    int m = L & 15, half = L >> 4;

    const unsigned short* arow = A + (unsigned)(r0 + m) * 64;
    v16bf a[2];
    #pragma unroll
    for (int c = 0; c < 2; c++) {
        const unsigned short* p = arow + 32 * c;
        a[c] = cat16(*reinterpret_cast<const v8bf*>(p + 8 * half),
                     *reinterpret_cast<const v8bf*>(p + 16 + 8 * half));
    }
    v8f acc[4];
    #pragma unroll
    for (int j = 0; j < 4; j++) acc[j] = splat8(bias ? bias[16 * j + m] : 0.f);

    #pragma unroll
    for (int c = 0; c < 2; c++) {
        #pragma unroll
        for (int j = 0; j < 4; j++) {
            const unsigned short* bp = Bt + (unsigned)(16 * j + m) * 64 + 32 * c + 8 * half;
            v16bf b = cat16(*reinterpret_cast<const v8bf*>(bp),
                            *reinterpret_cast<const v8bf*>(bp + 16));
            acc[j] = __builtin_amdgcn_wmma_f32_16x16x32_bf16(false, a[c], false, b,
                                                             (short)0, acc[j], false, false);
        }
    }
    #pragma unroll
    for (int j = 0; j < 4; j++) {
        int n = 16 * j + m;
        #pragma unroll
        for (int r = 0; r < 8; r++) {
            float v = acc[j][r];
            if (relu) v = fmaxf(v, 0.f);
            unsigned row = (unsigned)(r0 + r + 8 * half);
            if (Cf) Cf[row * 64 + n] = v;
            if (Cb) Cb[row * 64 + n] = f2bf(v);
        }
    }
}

// ---------------- graph-norm reductions ----------------
__global__ void colsum_k(const float* __restrict__ y, float* __restrict__ colsum, int n) {
    int col = threadIdx.x & 63;
    int rpb = blockDim.x / 64;
    float s = 0.f;
    for (int r = blockIdx.x * rpb + threadIdx.x / 64; r < n; r += gridDim.x * rpb)
        s += y[(unsigned)r * 64 + col];
    atomicAdd(&colsum[col], s);
}

__global__ void colvar_k(const float* __restrict__ y, const float* __restrict__ mu,
                         const float* __restrict__ sc, float* __restrict__ colsum2, int n) {
    int col = threadIdx.x & 63;
    int rpb = blockDim.x / 64;
    float mc = mu[col], scc = sc[col];
    float s = 0.f;
    for (int r = blockIdx.x * rpb + threadIdx.x / 64; r < n; r += gridDim.x * rpb) {
        float c = y[(unsigned)r * 64 + col] - scc * mc;
        s += c * c;
    }
    atomicAdd(&colsum2[col], s);
}

__global__ void mu_k(const float* colsum, float* mu, int n) {
    int i = threadIdx.x;
    if (i < 64) mu[i] = colsum[i] / (float)n;
}

__global__ void inv_k(const float* colsum2, const float* gnw, float* inv, int n) {
    int i = threadIdx.x;
    if (i < 64) inv[i] = gnw[i] * rsqrtf(colsum2[i] / (float)n + 1e-5f);
}

__global__ void norm_k(const float* __restrict__ y, const float* __restrict__ mu,
                       const float* __restrict__ sc, const float* __restrict__ inv,
                       const float* __restrict__ gnb, const float* __restrict__ res,
                       float* __restrict__ hn, unsigned short* __restrict__ hbn, int n) {
    int i = blockIdx.x * blockDim.x + threadIdx.x;
    if (i >= n) return;
    int f = i & 63;
    float c = y[i] - sc[f] * mu[f];
    float v = fmaxf(c * inv[f] + gnb[f], 0.f);
    float o = v + res[i];
    hn[i] = o;
    hbn[i] = f2bf(o);
}

__global__ void fc2_k(const float* __restrict__ z, const float* __restrict__ W,
                      const float* __restrict__ b, float* __restrict__ out, int n) {
    int i = blockIdx.x * blockDim.x + threadIdx.x;
    if (i >= n) return;
    const float* zr = z + (unsigned)i * 64;
    float s0 = b[0], s1 = b[1];
    #pragma unroll 8
    for (int k = 0; k < 64; k++) {
        float v = zr[k];
        s0 += v * W[k * 2 + 0];
        s1 += v * W[k * 2 + 1];
    }
    out[(unsigned)i * 2]     = s0;
    out[(unsigned)i * 2 + 1] = s1;
}

// ---------------- host orchestration ----------------

extern "C" void kernel_launch(void* const* d_in, const int* in_sizes, int n_in,
                              void* d_out, int out_size, void* d_ws, size_t ws_size,
                              hipStream_t stream) {
    const float* x      = (const float*)d_in[0];
    const int*   ei     = (const int*)d_in[1];
    const float* deg    = (const float*)d_in[2];
    const float* W_pre  = (const float*)d_in[3];
    const float* b_pre  = (const float*)d_in[4];
    const float* W_post = (const float*)d_in[5];
    const float* b_post = (const float*)d_in[6];
    const float* W_lin  = (const float*)d_in[7];
    const float* b_lin  = (const float*)d_in[8];
    const float* gnw    = (const float*)d_in[9];
    const float* gnb    = (const float*)d_in[10];
    const float* gns    = (const float*)d_in[11];
    const float* W_proj = (const float*)d_in[12];
    const float* W_fc1  = (const float*)d_in[13];
    const float* b_fc1  = (const float*)d_in[14];
    const float* W_fc2  = (const float*)d_in[15];
    const float* b_fc2  = (const float*)d_in[16];

    const int* src = ei;
    const int* tgt = ei + N_EDGES;

    char* cur = (char*)d_ws;
    auto alloc = [&](size_t bytes) -> void* {
        void* q = (void*)cur;
        cur += (bytes + 255) & ~(size_t)255;
        return q;
    };

    const size_t NF  = (size_t)N_NODES * FDIM;
    const size_t NTF = (size_t)N_NODES * TOW * FDIM;

    float* scalars = (float*)alloc(256);
    float* cnt     = (float*)alloc(N_NODES * 4);
    float* h0      = (float*)alloc(NF * 4);
    float* h1      = (float*)alloc(NF * 4);
    unsigned short* hb = (unsigned short*)alloc(NF * 2);
    float* hproj   = (float*)alloc(NF * 4);
    float* ybuf    = (float*)alloc(NF * 4);
    float* zbuf    = (float*)alloc(NF * 4);
    unsigned short* pb = (unsigned short*)alloc(NF * 2);
    float* sum  = (float*)alloc(NTF * 4);
    float* sum2 = (float*)alloc(NTF * 4);
    float* mnb  = (float*)alloc(NTF * 4);
    float* mxb  = (float*)alloc(NTF * 4);
    unsigned short* wpreT  = (unsigned short*)alloc(2 * 2 * 64 * 128 * 2);
    unsigned short* wpostT = (unsigned short*)alloc(2 * 2 * 32 * 832 * 2);
    unsigned short* wlinT  = (unsigned short*)alloc(2 * 64 * 64 * 2);
    unsigned short* wprojT = (unsigned short*)alloc(64 * 64 * 2);
    unsigned short* wfc1T  = (unsigned short*)alloc(64 * 64 * 2);
    float* colsum  = (float*)alloc(64 * 4);
    float* colsum2 = (float*)alloc(64 * 4);
    float* mubuf   = (float*)alloc(64 * 4);
    float* invbuf  = (float*)alloc(64 * 4);

    // prep
    avg_log_k<<<1, 64, 0, stream>>>(deg, scalars);
    fill_k<<<(N_NODES + 255) / 256, 256, 0, stream>>>(cnt, 0.f, N_NODES);
    count_k<<<(N_EDGES + 255) / 256, 256, 0, stream>>>(tgt, cnt, N_EDGES);

    for (int lt = 0; lt < 4; lt++) {
        transpose_bf16_k<<<(128 * 64 + 255) / 256, 256, 0, stream>>>(
            wpreT + (size_t)lt * 64 * 128, W_pre + (size_t)lt * 128 * 64, 128, 64);
        transpose_bf16_k<<<(832 * 32 + 255) / 256, 256, 0, stream>>>(
            wpostT + (size_t)lt * 32 * 832, W_post + (size_t)lt * 832 * 32, 832, 32);
    }
    for (int l = 0; l < NLAYER; l++)
        transpose_bf16_k<<<(64 * 64 + 255) / 256, 256, 0, stream>>>(
            wlinT + (size_t)l * 4096, W_lin + (size_t)l * 4096, 64, 64);
    transpose_bf16_k<<<(64 * 64 + 255) / 256, 256, 0, stream>>>(wprojT, W_proj, 64, 64);
    transpose_bf16_k<<<(64 * 64 + 255) / 256, 256, 0, stream>>>(wfc1T, W_fc1, 64, 64);

    (void)hipMemcpyAsync(h0, x, NF * 4, hipMemcpyDeviceToDevice, stream);
    convert_bf16_k<<<(int)((NF + 255) / 256), 256, 0, stream>>>(hb, x, (int)NF);

    // layer-0 residual projection: hproj = x @ W_proj
    gemm64_wmma_k<<<(N_NODES / 16 + 3) / 4, 128, 0, stream>>>(
        hb, wprojT, nullptr, hproj, nullptr, N_NODES, 0);

    float* hcur = h0;
    float* hnext = h1;
    for (int l = 0; l < NLAYER; l++) {
        init_acc_k<<<(int)((NTF + 255) / 256), 256, 0, stream>>>(sum, sum2, mnb, mxb, (int)NTF);

        pna_edge_k<<<N_EDGES / (16 * 4), 128, 0, stream>>>(
            hb, src, tgt,
            wpreT + (size_t)l * 2 * 64 * 128, b_pre + (size_t)l * 2 * 64,
            sum, sum2, mnb, mxb, N_EDGES);

        pna_finalize_k<<<N_NODES / 16, 32, 0, stream>>>(
            hcur, cnt, scalars, sum, sum2, mnb, mxb,
            wpostT + (size_t)l * 2 * 32 * 832, b_post + (size_t)l * 2 * 32, pb);

        gemm64_wmma_k<<<(N_NODES / 16 + 3) / 4, 128, 0, stream>>>(
            pb, wlinT + (size_t)l * 4096, b_lin + (size_t)l * 64, ybuf, nullptr, N_NODES, 0);

        fill_k<<<1, 64, 0, stream>>>(colsum, 0.f, 64);
        colsum_k<<<256, 256, 0, stream>>>(ybuf, colsum, N_NODES);
        mu_k<<<1, 64, 0, stream>>>(colsum, mubuf, N_NODES);
        fill_k<<<1, 64, 0, stream>>>(colsum2, 0.f, 64);
        colvar_k<<<256, 256, 0, stream>>>(ybuf, mubuf, gns + (size_t)l * 64, colsum2, N_NODES);
        inv_k<<<1, 64, 0, stream>>>(colsum2, gnw + (size_t)l * 64, invbuf, N_NODES);

        norm_k<<<(int)((NF + 255) / 256), 256, 0, stream>>>(
            ybuf, mubuf, gns + (size_t)l * 64, invbuf, gnb + (size_t)l * 64,
            (l == 0) ? hproj : hcur, hnext, hb, (int)NF);

        float* tmp = hcur; hcur = hnext; hnext = tmp;
    }

    // head: z = relu(h @ W_fc1 + b_fc1); out = z @ W_fc2 + b_fc2
    gemm64_wmma_k<<<(N_NODES / 16 + 3) / 4, 128, 0, stream>>>(
        hb, wfc1T, b_fc1, zbuf, nullptr, N_NODES, 1);
    fc2_k<<<(N_NODES + 255) / 256, 256, 0, stream>>>(zbuf, W_fc2, b_fc2, (float*)d_out, N_NODES);
}
